// SimpleMambaSSM_87832081203336
// MI455X (gfx1250) — matrix operand order repaired and checked by reading, not compiled, not run on hardware
//
#include <hip/hip_runtime.h>
#include <hip/hip_bf16.h>
#include <math.h>

// ---------------------------------------------------------------------------
// MI455X (gfx1250) fused Mamba-SSM block.
//   GEMM1/GEMM2 : split-bf16 (3x v_wmma_f32_16x16x32_bf16), fp32 accum,
//                 software-pipelined global->reg->LDS staging
//   scan        : chunk-parallel linear recurrence (local scan + carry
//                 propagation + correction), conv+SiLU fused into pass A
//   residual+LN : one memory pass
// ---------------------------------------------------------------------------

typedef __bf16 bf16;
typedef __attribute__((ext_vector_type(16))) __bf16 v16bf;
typedef __attribute__((ext_vector_type(8)))  __bf16 v8bf;
typedef __attribute__((ext_vector_type(4)))  __bf16 v4bf;
typedef __attribute__((ext_vector_type(8)))  float  v8f;

#define BATCH   4
#define SEQ     2048
#define D_MODEL 1024
#define D_INNER 2048
#define D_STATE 16
#define KCONV   4

#define NCHUNK  8
#define LCHUNK  (SEQ / NCHUNK)     // 256

// ---- GEMM tiling ----------------------------------------------------------
#define BM 128
#define BN 128
#define BK 32
#define LDST 40          // padded bf16 row stride (40*2B = 80B, 16B aligned)

__device__ __forceinline__ void split_bf16(float f, bf16 &hi, bf16 &lo) {
    hi = (bf16)f;
    lo = (bf16)(f - (float)hi);
}

__device__ __forceinline__ v16bf cat8(v8bf a, v8bf b) {
    return __builtin_shufflevector(a, b, 0,1,2,3,4,5,6,7,8,9,10,11,12,13,14,15);
}

// C[M,N] = A[M,K] * B[N,K]^T, fp32 in global, K-contiguous.
// fp32 operands split to (hi,lo) bf16 at LDS staging; 3 WMMA per tile-pair.
// Next K-tile is prefetched into registers while WMMAs run (latency hiding).
__global__ __launch_bounds__(256)
void gemm_f32_via_bf16x3(const float* __restrict__ A,
                         const float* __restrict__ B,
                         float* __restrict__ C,
                         int M, int N, int K) {
    __shared__ bf16 Ah[BM][LDST];
    __shared__ bf16 Al[BM][LDST];
    __shared__ bf16 Bh[BN][LDST];
    __shared__ bf16 Bl[BN][LDST];

    const int tid  = threadIdx.x;
    const int lane = tid & 31;
    const int wave = tid >> 5;
    const int wm   = wave >> 1;      // 0..3 : wave M position (32 rows each)
    const int wn   = wave & 1;       // 0..1 : wave N position (64 cols each)
    const int half = lane >> 4;
    const int l16  = lane & 15;

    const int m0 = blockIdx.y * BM;
    const int n0 = blockIdx.x * BN;

    // fixed per-thread staging coordinates (4 float4 of A and of B per step)
    int rj[4], cj[4];
#pragma unroll
    for (int j = 0; j < 4; ++j) {
        const int f4 = tid + j * 256;      // 0..1023, 8 float4 per 32-wide row
        rj[j] = f4 >> 3;
        cj[j] = (f4 & 7) << 2;
    }

    v8f acc[2][4];
#pragma unroll
    for (int mt = 0; mt < 2; ++mt)
#pragma unroll
        for (int nt = 0; nt < 4; ++nt)
            acc[mt][nt] = (v8f){0.f,0.f,0.f,0.f,0.f,0.f,0.f,0.f};

    // ---- prologue: load K-tile 0 into registers ---------------------------
    float4 aR[4], bR[4];
#pragma unroll
    for (int j = 0; j < 4; ++j) {
        aR[j] = *(const float4*)(A + (size_t)(m0 + rj[j]) * K + cj[j]);
        bR[j] = *(const float4*)(B + (size_t)(n0 + rj[j]) * K + cj[j]);
    }

    for (int kb = 0; kb < K; kb += BK) {
        // ---- store current tile (registers) to LDS as hi/lo bf16 ----------
        __syncthreads();
#pragma unroll
        for (int j = 0; j < 4; ++j) {
            bf16 h0,h1,h2,h3,l0,l1,l2,l3;
            split_bf16(aR[j].x,h0,l0); split_bf16(aR[j].y,h1,l1);
            split_bf16(aR[j].z,h2,l2); split_bf16(aR[j].w,h3,l3);
            *(v4bf*)&Ah[rj[j]][cj[j]] = (v4bf){h0,h1,h2,h3};
            *(v4bf*)&Al[rj[j]][cj[j]] = (v4bf){l0,l1,l2,l3};
            split_bf16(bR[j].x,h0,l0); split_bf16(bR[j].y,h1,l1);
            split_bf16(bR[j].z,h2,l2); split_bf16(bR[j].w,h3,l3);
            *(v4bf*)&Bh[rj[j]][cj[j]] = (v4bf){h0,h1,h2,h3};
            *(v4bf*)&Bl[rj[j]][cj[j]] = (v4bf){l0,l1,l2,l3};
        }
        __syncthreads();

        // ---- prefetch next K-tile into registers (hidden behind WMMAs) ----
        const int  kn   = kb + BK;
        const bool more = kn < K;
        float4 aN[4], bN[4];
        if (more) {
#pragma unroll
            for (int j = 0; j < 4; ++j) {
                aN[j] = *(const float4*)(A + (size_t)(m0 + rj[j]) * K + kn + cj[j]);
                bN[j] = *(const float4*)(B + (size_t)(n0 + rj[j]) * K + kn + cj[j]);
            }
        }

        // ---- A fragments: 16x32 bf16 (ISA 7.12.2 layout) ------------------
        v16bf afh[2], afl[2];
#pragma unroll
        for (int mt = 0; mt < 2; ++mt) {
            const int ar = wm * 32 + mt * 16 + l16;
            afh[mt] = cat8(*(const v8bf*)&Ah[ar][half * 8],
                           *(const v8bf*)&Ah[ar][16 + half * 8]);
            afl[mt] = cat8(*(const v8bf*)&Al[ar][half * 8],
                           *(const v8bf*)&Al[ar][16 + half * 8]);
        }
        // ---- B fragments (32x16) + WMMA -----------------------------------
#pragma unroll
        for (int nt = 0; nt < 4; ++nt) {
            const int bc = wn * 64 + nt * 16 + l16;
            v16bf bfh = cat8(*(const v8bf*)&Bh[bc][half * 8],
                             *(const v8bf*)&Bh[bc][16 + half * 8]);
            v16bf bfl = cat8(*(const v8bf*)&Bl[bc][half * 8],
                             *(const v8bf*)&Bl[bc][16 + half * 8]);
#pragma unroll
            for (int mt = 0; mt < 2; ++mt) {
                // hi*hi + hi*lo + lo*hi : ~2^-16 relative error, fp32 accum
                acc[mt][nt] = __builtin_amdgcn_wmma_f32_16x16x32_bf16(
                    false, afh[mt], false, bfh, (short)0, acc[mt][nt], false, false);
                acc[mt][nt] = __builtin_amdgcn_wmma_f32_16x16x32_bf16(
                    false, afh[mt], false, bfl, (short)0, acc[mt][nt], false, false);
                acc[mt][nt] = __builtin_amdgcn_wmma_f32_16x16x32_bf16(
                    false, afl[mt], false, bfh, (short)0, acc[mt][nt], false, false);
            }
        }

        if (more) {
#pragma unroll
            for (int j = 0; j < 4; ++j) { aR[j] = aN[j]; bR[j] = bN[j]; }
        }
    }

    // ---- epilogue: C v8f layout: elem r -> row r + 8*half, col l16 --------
#pragma unroll
    for (int mt = 0; mt < 2; ++mt)
#pragma unroll
        for (int nt = 0; nt < 4; ++nt) {
            const int gcol = n0 + wn * 64 + nt * 16 + l16;
#pragma unroll
            for (int r = 0; r < 8; ++r) {
                const int grow = m0 + wm * 32 + mt * 16 + half * 8 + r;
                C[(size_t)grow * N + gcol] = acc[mt][nt][r];
            }
        }
}

// ---------------------------------------------------------------------------
// Pass A: chunk-local conv + SiLU gate + scan from zero state.
// One thread per (b, chunk, d): 65536 threads = 2048 waves.
// Writes provisional ys and the chunk's final local state.
// ---------------------------------------------------------------------------
__global__ __launch_bounds__(256)
void scan_chunk_local(const float* __restrict__ xp,
                      const float* __restrict__ Am, const float* __restrict__ Bm,
                      const float* __restrict__ Cm, const float* __restrict__ Dv,
                      const float* __restrict__ convw, const float* __restrict__ convb,
                      float* __restrict__ ys, float* __restrict__ lf) {
    const int g  = blockIdx.x * blockDim.x + threadIdx.x;
    const int d  = g & (D_INNER - 1);
    const int bk = g >> 11;                    // 0..31
    const int k  = bk & (NCHUNK - 1);
    const int b  = bk >> 3;
    const int t0 = k * LCHUNK;

    float a[D_STATE], bb[D_STATE], cc[D_STATE], st[D_STATE];
#pragma unroll
    for (int s = 0; s < D_STATE; ++s) {
        a[s]  = Am[d * D_STATE + s];
        bb[s] = Bm[d * D_STATE + s];
        cc[s] = Cm[d * D_STATE + s];
        st[s] = 0.f;
    }
    const float w0 = convw[d * KCONV + 0], w1 = convw[d * KCONV + 1];
    const float w2 = convw[d * KCONV + 2], w3 = convw[d * KCONV + 3];
    const float cb = convb[d];
    const float Dd = Dv[d];

    const float* xrow = xp + (size_t)b * SEQ * 2 * D_INNER + d;
    float*       yrow = ys + (size_t)b * SEQ * D_INNER + d;

    // conv taps are pure functions of xp: pull window across chunk boundary
    float xm3 = 0.f, xm2 = 0.f, xm1 = 0.f;
    if (k > 0) {
        xm3 = xrow[(size_t)(t0 - 3) * 2 * D_INNER];
        xm2 = xrow[(size_t)(t0 - 2) * 2 * D_INNER];
        xm1 = xrow[(size_t)(t0 - 1) * 2 * D_INNER];
    }

    for (int t = t0; t < t0 + LCHUNK; ++t) {
        const float u  = xrow[(size_t)t * 2 * D_INNER];
        const float gt = xrow[(size_t)t * 2 * D_INNER + D_INNER];
        float xc = fmaf(w0, xm3, fmaf(w1, xm2, fmaf(w2, xm1, fmaf(w3, u, cb))));
        xm3 = xm2; xm2 = xm1; xm1 = u;
        const float sig = 1.f / (1.f + __expf(-gt));
        xc *= gt * sig;
        float out = Dd * xc;
#pragma unroll
        for (int s = 0; s < D_STATE; ++s) {
            st[s] = fmaf(st[s], a[s], xc * bb[s]);
            out   = fmaf(st[s], cc[s], out);
        }
        yrow[(size_t)t * D_INNER] = out;
    }
    // final local state: layout [b][k][s][d] for coalesced pass B/C reads
#pragma unroll
    for (int s = 0; s < D_STATE; ++s)
        lf[(((size_t)(b * NCHUNK + k)) * D_STATE + s) * D_INNER + d] = st[s];
}

// ---------------------------------------------------------------------------
// Pass B: sequential carry propagation across the 8 chunks (tiny).
// One thread per (b, d). carry_{k} = a^L (.) carry_{k-1} + localfinal_{k-1}.
// ---------------------------------------------------------------------------
__global__ __launch_bounds__(256)
void scan_carry(const float* __restrict__ Am,
                const float* __restrict__ lf, float* __restrict__ ci) {
    const int g = blockIdx.x * blockDim.x + threadIdx.x;
    const int d = g & (D_INNER - 1);
    const int b = g >> 11;
    if (b >= BATCH) return;

    float aL[D_STATE], carry[D_STATE];
#pragma unroll
    for (int s = 0; s < D_STATE; ++s) {
        float p = Am[d * D_STATE + s];
#pragma unroll
        for (int q = 0; q < 8; ++q) p = p * p;   // a^(2^8) = a^LCHUNK
        aL[s]    = p;
        carry[s] = 0.f;
    }
    for (int k = 0; k < NCHUNK; ++k) {
#pragma unroll
        for (int s = 0; s < D_STATE; ++s) {
            const size_t idx = (((size_t)(b * NCHUNK + k)) * D_STATE + s) * D_INNER + d;
            ci[idx]  = carry[s];
            carry[s] = fmaf(aL[s], carry[s], lf[idx]);
        }
    }
}

// ---------------------------------------------------------------------------
// Pass C: correction. True state = a^(t-t0+1) (.) carry + local, so
// ys[t] += sum_s c_s * a_s^(t-t0+1) * carry_s.  k==0 has zero carry -> skip.
// ---------------------------------------------------------------------------
__global__ __launch_bounds__(256)
void scan_fixup(const float* __restrict__ Am, const float* __restrict__ Cm,
                const float* __restrict__ ci, float* __restrict__ ys) {
    const int g  = blockIdx.x * blockDim.x + threadIdx.x;
    const int d  = g & (D_INNER - 1);
    const int bk = g >> 11;
    const int k  = bk & (NCHUNK - 1);
    const int b  = bk >> 3;
    if (k == 0) return;                        // wave-uniform exit
    const int t0 = k * LCHUNK;

    float a[D_STATE], cc[D_STATE], w[D_STATE];
#pragma unroll
    for (int s = 0; s < D_STATE; ++s) {
        a[s]  = Am[d * D_STATE + s];
        cc[s] = Cm[d * D_STATE + s];
        w[s]  = ci[(((size_t)(b * NCHUNK + k)) * D_STATE + s) * D_INNER + d] * a[s];
    }
    float* yrow = ys + (size_t)b * SEQ * D_INNER + d;
    for (int t = t0; t < t0 + LCHUNK; ++t) {
        float add = 0.f;
#pragma unroll
        for (int s = 0; s < D_STATE; ++s) {
            add  = fmaf(w[s], cc[s], add);
            w[s] *= a[s];
        }
        yrow[(size_t)t * D_INNER] += add;
    }
}

// ---------------------------------------------------------------------------
// residual + LayerNorm: one block per (b,s) row of 1024.
// ---------------------------------------------------------------------------
__global__ __launch_bounds__(256)
void residual_layernorm(const float* __restrict__ y, const float* __restrict__ x,
                        const float* __restrict__ gam, const float* __restrict__ bet,
                        float* __restrict__ out) {
    __shared__ float rs[256];
    __shared__ float rq[256];
    const int tid = threadIdx.x;
    const size_t row = blockIdx.x;

    const float4 yv = ((const float4*)(y + row * D_MODEL))[tid];
    const float4 xv = ((const float4*)(x + row * D_MODEL))[tid];
    const float z0 = yv.x + xv.x, z1 = yv.y + xv.y;
    const float z2 = yv.z + xv.z, z3 = yv.w + xv.w;

    rs[tid] = z0 + z1 + z2 + z3;
    rq[tid] = z0*z0 + z1*z1 + z2*z2 + z3*z3;
    __syncthreads();
#pragma unroll
    for (int off = 128; off > 0; off >>= 1) {
        if (tid < off) { rs[tid] += rs[tid + off]; rq[tid] += rq[tid + off]; }
        __syncthreads();
    }
    const float mu  = rs[0] * (1.f / D_MODEL);
    const float var = rq[0] * (1.f / D_MODEL) - mu * mu;
    const float inv = rsqrtf(var + 1e-5f);

    const float4 gv = ((const float4*)gam)[tid];
    const float4 bv = ((const float4*)bet)[tid];
    float4 ov;
    ov.x = (z0 - mu) * inv * gv.x + bv.x;
    ov.y = (z1 - mu) * inv * gv.y + bv.y;
    ov.z = (z2 - mu) * inv * gv.z + bv.z;
    ov.w = (z3 - mu) * inv * gv.w + bv.w;
    ((float4*)(out + row * D_MODEL))[tid] = ov;
}

// ---------------------------------------------------------------------------
extern "C" void kernel_launch(void* const* d_in, const int* in_sizes, int n_in,
                              void* d_out, int out_size, void* d_ws, size_t ws_size,
                              hipStream_t stream) {
    const float* x     = (const float*)d_in[0];
    const float* W_in  = (const float*)d_in[1];
    const float* W_out = (const float*)d_in[2];
    const float* Am    = (const float*)d_in[3];
    const float* Bm    = (const float*)d_in[4];
    const float* Cm    = (const float*)d_in[5];
    const float* Dv    = (const float*)d_in[6];
    const float* convw = (const float*)d_in[7];
    const float* convb = (const float*)d_in[8];
    const float* lng   = (const float*)d_in[9];
    const float* lnb   = (const float*)d_in[10];
    float* out = (float*)d_out;

    const int M = BATCH * SEQ;                 // 8192 rows
    float* ws = (float*)d_ws;
    float* xp = ws;                            // (M, 2*D_INNER)   128 MB
    float* ys = xp + (size_t)M * 2 * D_INNER;  // (M, D_INNER)      64 MB
    float* yb = ys + (size_t)M * D_INNER;      // (M, D_MODEL)      32 MB
    float* lf = yb + (size_t)M * D_MODEL;      // chunk final states  4 MB
    float* ci = lf + (size_t)BATCH * NCHUNK * D_STATE * D_INNER;  //    4 MB

    // GEMM1: xp = x @ W_in^T  (M=8192, N=4096, K=1024)
    gemm_f32_via_bf16x3<<<dim3((2 * D_INNER) / BN, M / BM), 256, 0, stream>>>(
        x, W_in, xp, M, 2 * D_INNER, D_MODEL);

    // chunk-parallel scan: local pass, carry propagation, correction
    scan_chunk_local<<<(BATCH * NCHUNK * D_INNER) / 256, 256, 0, stream>>>(
        xp, Am, Bm, Cm, Dv, convw, convb, ys, lf);
    scan_carry<<<(BATCH * D_INNER) / 256, 256, 0, stream>>>(Am, lf, ci);
    scan_fixup<<<(BATCH * NCHUNK * D_INNER) / 256, 256, 0, stream>>>(Am, Cm, ci, ys);

    // GEMM2: yb = ys @ W_out^T (M=8192, N=1024, K=2048)
    gemm_f32_via_bf16x3<<<dim3(D_MODEL / BN, M / BM), 256, 0, stream>>>(
        ys, W_out, yb, M, D_MODEL, D_INNER);

    // residual + layernorm -> d_out
    residual_layernorm<<<M, 256, 0, stream>>>(yb, x, lng, lnb, out);
}
